// MultiHeadAttention_48017734369716
// MI455X (gfx1250) — matrix operand order repaired and checked by reading, not compiled, not run on hardware
//
#include <hip/hip_runtime.h>

// ---------------------------------------------------------------------------
// MHA forward for MI455X (gfx1250, wave32, WMMA bf16 16x16x32, async-to-LDS)
// ---------------------------------------------------------------------------

typedef __bf16 bf16_t;
typedef bf16_t v8bf  __attribute__((ext_vector_type(8)));
typedef bf16_t v16bf __attribute__((ext_vector_type(16)));
typedef float  v8f   __attribute__((ext_vector_type(8)));

#define BATCH 4
#define SEQ   2048
#define EMB   768
#define NH    12
#define HD    64
#define ROWS  (BATCH * SEQ)      // 8192
#define QKV_N (3 * EMB)          // 2304

// 1/sqrt(64) * log2(e): scores come out pre-scaled for exp2-based softmax
#define QSCALE (0.125f * 1.44269504088896340736f)

// ---------------------------------------------------------------------------
// helpers
// ---------------------------------------------------------------------------

__device__ __forceinline__ v16bf ld_pair(const bf16_t* p0, const bf16_t* p1) {
    v8bf lo = *(const v8bf*)p0;
    v8bf hi = *(const v8bf*)p1;
    v16bf r;
#pragma unroll
    for (int i = 0; i < 8; ++i) { r[i] = lo[i]; r[i + 8] = hi[i]; }
    return r;
}

// A-operand (16x32 bf16, row-major source, leading dim ld):
// lane 0-15 holds row M=lane, K = {c..c+7, 16+c..16+c+7} with c = (lane>=16)*8
__device__ __forceinline__ v16bf ldA(const bf16_t* base, int ld, int row0, int k0, int lane) {
    int r = row0 + (lane & 15);
    int c = k0 + ((lane >> 4) << 3);
    const bf16_t* p = base + (size_t)r * ld + c;
    return ld_pair(p, p + 16);
}

// B-operand (32x16 bf16, row-major K x N source): lane holds column N = n0+lane%16,
// K = {c..c+7, 16+c..16+c+7} with c = k0 + (lane>=16)*8
__device__ __forceinline__ v16bf ldB(const bf16_t* base, int ld, int k0, int n0, int lane) {
    int n = n0 + (lane & 15);
    int c = k0 + ((lane >> 4) << 3);
    v16bf r;
#pragma unroll
    for (int i = 0; i < 8; ++i) {
        r[i]     = base[(size_t)(c + i) * ld + n];
        r[i + 8] = base[(size_t)(c + 16 + i) * ld + n];
    }
    return r;
}

__device__ __forceinline__ v8f wmma_bf16(v16bf a, v16bf b, v8f c) {
    return __builtin_amdgcn_wmma_f32_16x16x32_bf16(
        /*neg_a=*/false, a, /*neg_b=*/false, b,
        /*c_mod=*/(short)0, c, /*reuse_a=*/false, /*reuse_b=*/false);
}

// reductions over the 16-lane half (xor masks 1,2,4,8 stay within each half)
__device__ __forceinline__ float redmax16(float x) {
    x = fmaxf(x, __shfl_xor(x, 1));
    x = fmaxf(x, __shfl_xor(x, 2));
    x = fmaxf(x, __shfl_xor(x, 4));
    x = fmaxf(x, __shfl_xor(x, 8));
    return x;
}
__device__ __forceinline__ float redsum16(float x) {
    x += __shfl_xor(x, 1);
    x += __shfl_xor(x, 2);
    x += __shfl_xor(x, 4);
    x += __shfl_xor(x, 8);
    return x;
}

// async global->LDS copy of 16 bytes (CDNA5 ASYNCcnt path)
__device__ __forceinline__ void async_b128(unsigned lds_addr, const void* gptr) {
    unsigned long long ga = (unsigned long long)(uintptr_t)gptr;
    asm volatile("global_load_async_to_lds_b128 %0, %1, off"
                 :: "v"(lds_addr), "v"(ga) : "memory");
}
__device__ __forceinline__ void wait_async0() {
    asm volatile("s_wait_asynccnt 0" ::: "memory");
}

// ---------------------------------------------------------------------------
// K0: fp32 -> bf16 conversion
// ---------------------------------------------------------------------------
__global__ void cvt_f32_bf16(const float* __restrict__ src, bf16_t* __restrict__ dst, int n) {
    int i = blockIdx.x * blockDim.x + threadIdx.x;
    if (i < n) dst[i] = (bf16_t)src[i];
}

// ---------------------------------------------------------------------------
// K1: QKV GEMM  [8192 x 768] x [768 x 2304] -> scatter into Q/K/V [B,H,S,D]
//     one wave computes a 32x64 output tile (2 M x 4 N), fp32 accum, bf16 out
// ---------------------------------------------------------------------------
__global__ void __launch_bounds__(256)
qkv_gemm(const bf16_t* __restrict__ xh, const bf16_t* __restrict__ w,
         bf16_t* __restrict__ Qh, bf16_t* __restrict__ Kh, bf16_t* __restrict__ Vh) {
    int lane = threadIdx.x & 31;
    int wid  = blockIdx.x * 8 + (threadIdx.x >> 5);
    int rowTile = wid & 255;      // 256 row tiles of 32
    int colTile = wid >> 8;       // 36 col tiles of 64
    int row0 = rowTile << 5;
    int col0 = colTile << 6;

    v8f acc[2][4] = {};
#pragma unroll 2
    for (int k0 = 0; k0 < EMB; k0 += 32) {
        v16bf a0 = ldA(xh, EMB, row0,      k0, lane);
        v16bf a1 = ldA(xh, EMB, row0 + 16, k0, lane);
#pragma unroll
        for (int t = 0; t < 4; ++t) {
            v16bf b = ldB(w, QKV_N, k0, col0 + 16 * t, lane);
            acc[0][t] = wmma_bf16(a0, b, acc[0][t]);
            acc[1][t] = wmma_bf16(a1, b, acc[1][t]);
        }
    }

    int half = lane >> 4;
    int nl   = lane & 15;
#pragma unroll
    for (int t = 0; t < 4; ++t) {
        int c = col0 + t * 16 + nl;      // column in [0, 2304)
        int which = c / EMB;             // 0=Q 1=K 2=V
        int e = c % EMB;
        int h = e >> 6;
        int d = e & 63;
        bf16_t* dst = (which == 0) ? Qh : (which == 1) ? Kh : Vh;
        float scale = (which == 0) ? QSCALE : 1.0f;   // softmax scale folded into Q
#pragma unroll
        for (int mt = 0; mt < 2; ++mt)
#pragma unroll
            for (int v = 0; v < 8; ++v) {
                int s = row0 + mt * 16 + v + 8 * half;   // row in [0, 8192)
                int b = s >> 11;                         // batch
                int si = s & 2047;
                size_t idx = (((size_t)b * NH + h) * SEQ + si) * HD + d;
                dst[idx] = (bf16_t)(acc[mt][t][v] * scale);
            }
    }
}

// ---------------------------------------------------------------------------
// K2: flash attention. block = 8 waves sharing one (b,h); each wave owns 16
//     query rows; KV tiles of 32; K transposed in LDS; V tile async->LDS;
//     exp2-based online softmax (scale pre-folded into Q).
// ---------------------------------------------------------------------------
__global__ void __launch_bounds__(256)
attn(const bf16_t* __restrict__ Qh, const bf16_t* __restrict__ Kh,
     const bf16_t* __restrict__ Vh, bf16_t* __restrict__ Oh) {
    __shared__ __align__(64) bf16_t Kt[HD * 32];       // K tile transposed [d][kv], 4KB
    __shared__ __align__(64) bf16_t Vt[32 * HD];       // V tile row-major  [kv][d], 4KB
    __shared__ __align__(64) bf16_t Pb[8][16 * 32];    // per-wave P staging, 8KB

    int bh = blockIdx.x >> 4;           // 48 (b,h) pairs
    int qb = blockIdx.x & 15;           // 16 q-blocks of 128 rows
    int wv   = threadIdx.x >> 5;
    int lane = threadIdx.x & 31;
    int tid  = threadIdx.x;
    int q0 = qb * 128 + wv * 16;

    const bf16_t* Qp = Qh + (size_t)bh * SEQ * HD;
    const bf16_t* Kp = Kh + (size_t)bh * SEQ * HD;
    const bf16_t* Vp = Vh + (size_t)bh * SEQ * HD;

    // Q A-operand tiles, K-dim split d=[0,32) and [32,64)
    v16bf qa0 = ldA(Qp, HD, q0, 0, lane);
    v16bf qa1 = ldA(Qp, HD, q0, 32, lane);

    v8f o[4] = {};
    float m[8], l[8];
#pragma unroll
    for (int v = 0; v < 8; ++v) { m[v] = -__builtin_inff(); l[v] = 0.0f; }

    int half = lane >> 4;
    int nl   = lane & 15;
    bf16_t* pw = &Pb[wv][0];
    // generic LDS pointer low 32 bits == LDS byte offset (flat aperture rule)
    unsigned vtBase = (unsigned)(uintptr_t)&Vt[0];

    // cooperative tile indices: 256 threads, 16B each, over a 32x64 bf16 tile
    int cr = tid >> 3;            // row 0..31
    int cd = (tid & 7) << 3;      // col base 0..56

    for (int kv0 = 0; kv0 < SEQ; kv0 += 32) {
        __syncthreads();          // previous tile fully consumed

        // V tile: async DMA global -> LDS (row-major, no transpose needed)
        async_b128(vtBase + (unsigned)(cr * HD + cd) * 2,
                   Vp + (size_t)(kv0 + cr) * HD + cd);

        // K tile: load + transpose into LDS: Kt[d][kv]
        {
            v8bf kk = *(const v8bf*)(Kp + (size_t)(kv0 + cr) * HD + cd);
#pragma unroll
            for (int i = 0; i < 8; ++i) Kt[(cd + i) * 32 + cr] = kk[i];
        }

        // prefetch next KV tile while this one is being consumed
        int nkv = (kv0 + 32) & (SEQ - 1);
        __builtin_prefetch(Kp + (size_t)(nkv + cr) * HD + cd, 0, 1);
        __builtin_prefetch(Vp + (size_t)(nkv + cr) * HD + cd, 0, 1);

        wait_async0();            // this wave's async portion landed in LDS
        __syncthreads();          // all waves' K stores + V DMA visible

        // scores: S = Q * K^T  (two 16x16 N-subtiles, K-dim 64 = 2 chained WMMAs)
        v8f s0 = {}, s1 = {};
        s0 = wmma_bf16(qa0, ldB(Kt, 32, 0, 0, lane), s0);
        s0 = wmma_bf16(qa1, ldB(Kt, 32, 32, 0, lane), s0);
        s1 = wmma_bf16(qa0, ldB(Kt, 32, 0, 16, lane), s1);
        s1 = wmma_bf16(qa1, ldB(Kt, 32, 32, 16, lane), s1);

        // online softmax (base-2: scale already folded into Q)
        float alpha[8];
#pragma unroll
        for (int v = 0; v < 8; ++v) {
            float tm = redmax16(fmaxf(s0[v], s1[v]));
            float mn = fmaxf(m[v], tm);
            alpha[v] = exp2f(m[v] - mn);
            m[v] = mn;
        }
#pragma unroll
        for (int v = 0; v < 8; ++v) {
            s0[v] = exp2f(s0[v] - m[v]);
            s1[v] = exp2f(s1[v] - m[v]);
            float ts = redsum16(s0[v] + s1[v]);
            l[v] = l[v] * alpha[v] + ts;
        }
#pragma unroll
        for (int t = 0; t < 4; ++t)
#pragma unroll
            for (int v = 0; v < 8; ++v) o[t][v] *= alpha[v];

        // stage P (C-layout -> row-major bf16 in LDS -> A-layout reload)
#pragma unroll
        for (int v = 0; v < 8; ++v) {
            int r = v + 8 * half;
            pw[r * 32 + nl]      = (bf16_t)s0[v];
            pw[r * 32 + 16 + nl] = (bf16_t)s1[v];
        }
        asm volatile("s_wait_dscnt 0" ::: "memory");
        v16bf pa = ldA(pw, 32, 0, 0, lane);

        // O += P * V  (V B-operand from the async-loaded LDS tile)
#pragma unroll
        for (int t = 0; t < 4; ++t) {
            v16bf bv = ldB(Vt, HD, 0, t * 16, lane);
            o[t] = wmma_bf16(pa, bv, o[t]);
        }
    }

    // epilogue: normalize and store to O [B, S, E] as bf16
    int b = bh / NH;
    int h = bh % NH;
#pragma unroll
    for (int v = 0; v < 8; ++v) {
        float inv = 1.0f / l[v];
        int s = q0 + v + 8 * half;
#pragma unroll
        for (int t = 0; t < 4; ++t) {
            size_t idx = ((size_t)b * SEQ + s) * EMB + h * HD + t * 16 + nl;
            Oh[idx] = (bf16_t)(o[t][v] * inv);
        }
    }
}

// ---------------------------------------------------------------------------
// K3: output projection [8192 x 768] x [768 x 768] + bias -> fp32 out
//     one wave computes a 32x64 tile (2 M x 4 N)
// ---------------------------------------------------------------------------
__global__ void __launch_bounds__(256)
proj_gemm(const bf16_t* __restrict__ Oh, const bf16_t* __restrict__ w,
          const float* __restrict__ bias, float* __restrict__ out) {
    int lane = threadIdx.x & 31;
    int wid  = blockIdx.x * 8 + (threadIdx.x >> 5);
    int rowTile = wid & 255;      // 256 row tiles of 32
    int colTile = wid >> 8;       // 12 col tiles of 64
    int row0 = rowTile << 5;
    int col0 = colTile << 6;

    v8f acc[2][4] = {};
#pragma unroll 2
    for (int k0 = 0; k0 < EMB; k0 += 32) {
        v16bf a0 = ldA(Oh, EMB, row0,      k0, lane);
        v16bf a1 = ldA(Oh, EMB, row0 + 16, k0, lane);
#pragma unroll
        for (int t = 0; t < 4; ++t) {
            v16bf b = ldB(w, EMB, k0, col0 + 16 * t, lane);
            acc[0][t] = wmma_bf16(a0, b, acc[0][t]);
            acc[1][t] = wmma_bf16(a1, b, acc[1][t]);
        }
    }

    int half = lane >> 4;
    int nl   = lane & 15;
#pragma unroll
    for (int t = 0; t < 4; ++t) {
        int c = col0 + t * 16 + nl;
        float bv = bias[c];
#pragma unroll
        for (int mt = 0; mt < 2; ++mt)
#pragma unroll
            for (int v = 0; v < 8; ++v) {
                int s = row0 + mt * 16 + v + 8 * half;
                out[(size_t)s * EMB + c] = acc[mt][t][v] + bv;
            }
    }
}

// ---------------------------------------------------------------------------
// host launcher
// ---------------------------------------------------------------------------
extern "C" void kernel_launch(void* const* d_in, const int* in_sizes, int n_in,
                              void* d_out, int out_size, void* d_ws, size_t ws_size,
                              hipStream_t stream) {
    const float* x      = (const float*)d_in[0];   // [4, 2048, 768]
    const float* W_qkv  = (const float*)d_in[1];   // [768, 2304]
    const float* W_proj = (const float*)d_in[2];   // [768, 768]
    const float* b_proj = (const float*)d_in[3];   // [768]
    float* out = (float*)d_out;                    // [4, 2048, 768]

    char* ws = (char*)d_ws;
    const size_t SZ_X    = (size_t)ROWS * EMB * sizeof(bf16_t);     // 12.6 MB
    const size_t SZ_WQKV = (size_t)EMB * QKV_N * sizeof(bf16_t);    // 3.5 MB
    const size_t SZ_WP   = (size_t)EMB * EMB * sizeof(bf16_t);      // 1.2 MB
    bf16_t* xh  = (bf16_t*)(ws);
    bf16_t* wq  = (bf16_t*)(ws + SZ_X);
    bf16_t* wp  = (bf16_t*)(ws + SZ_X + SZ_WQKV);
    bf16_t* Qh  = (bf16_t*)(ws + SZ_X + SZ_WQKV + SZ_WP);
    bf16_t* Kh  = (bf16_t*)(ws + 2 * SZ_X + SZ_WQKV + SZ_WP);
    bf16_t* Vh  = (bf16_t*)(ws + 3 * SZ_X + SZ_WQKV + SZ_WP);
    bf16_t* Oh  = (bf16_t*)(ws + 4 * SZ_X + SZ_WQKV + SZ_WP);

    // 1) fp32 -> bf16 conversions
    {
        int n = ROWS * EMB;
        cvt_f32_bf16<<<(n + 255) / 256, 256, 0, stream>>>(x, xh, n);
        n = EMB * QKV_N;
        cvt_f32_bf16<<<(n + 255) / 256, 256, 0, stream>>>(W_qkv, wq, n);
        n = EMB * EMB;
        cvt_f32_bf16<<<(n + 255) / 256, 256, 0, stream>>>(W_proj, wp, n);
    }

    // 2) QKV GEMM: 256 x 36 wave-tiles (32x64 each), 8 waves/block
    qkv_gemm<<<(256 * 36) / 8, 256, 0, stream>>>(xh, wq, Qh, Kh, Vh);

    // 3) flash attention: 48 (b,h) x 16 q-blocks -> 768 blocks of 8 waves
    attn<<<48 * 16, 256, 0, stream>>>(Qh, Kh, Vh, Oh);

    // 4) projection GEMM: 256 x 12 wave-tiles -> 384 blocks
    proj_gemm<<<(256 * 12) / 8, 256, 0, stream>>>(Oh, wp, b_proj, out);
}
